// DTP_21852793602298
// MI455X (gfx1250) — compile-verified, matrix operand order checked by reading, not run on hardware
//
#include <hip/hip_runtime.h>

// ---------------------------------------------------------------------------
// SE(3)-equivariant conv layer for MI455X (gfx1250, wave32, WMMA).
// Dominant bilinear  c[e,o] = sum_t sum_i h[e,t] * w3[t,o,i] * g[e,i]
// runs as per-node (16-edge) GEMMs on v_wmma_f32_16x16x32_bf16 with the
// radial w3 weights stationary in LDS in the WMMA-B lane layout (bf16).
// The sum_t contraction is fused in-register via immediate ds_swizzle
// butterfly reduction (8 sums over 16 lanes in 8 swizzles).
// ---------------------------------------------------------------------------

typedef __attribute__((ext_vector_type(16))) __bf16 v16bf;
typedef __attribute__((ext_vector_type(8)))  float  v8f;

#define N_      1024
#define NODES   2048      // B*N
#define NPB     8         // nodes per block
#define EPS_    1e-5f

// ---- LDS ushort (bf16) region offsets, in ushorts ----
#define U_WB00 0          // head00 w3 : 32 o * 2 kstep * 512
#define U_WB10 32768      // head10 w3 : 32 o * 512
#define U_WB01 49152      // head01 w3 : 16 o * 2 kstep * 512
#define U_WB11 65536      // head11 w3 (+b3 row @k=16) : 32 tiles * 512
#define U_BB00 81920      // head00 b3 tiles : 4 * 512
#define U_BB10 83968      // head10 b3 tiles : 2 * 512
#define U_BB01 84992      // head01 b3 tiles : 2 * 512
#define U_AG0  86016      // A operand: g0 (16x64) : 2 * 512
#define U_AG1  87040      // A operand: g1r[:, :, m=1] (16x32) : 512
#define U_AH11 87552      // A operand: h11 padded (16x32, k16=1.0) : 512
#define U_TOT  88064
#define U_BYTES (U_TOT * 2)          // 176128 B, 32B-aligned

// ---- LDS float region offsets, in floats ----
#define F_HB   0          // H multipliers, heads 00/10/01 : 3*16*16
#define F_TB   768        // basis-contracted t[e][i][m]  : 16*32*3
#define F_CB0  2304       // degree-0 edge results [o=64][e=16]
#define F_CB1  3328       // degree-1 edge results [l=32][a=3][e=16]
#define F_MSK  4864       // 16
#define F_D1   4880       // Wigner D per edge [e=16][9]
#define F_RD   5024       // rel_dist per edge : 16
#define F_P0   5040       // pooled deg0 : 64
#define F_P1   5104       // pooled deg1 : 96
#define F_DEN  5200       // 1
#define F_PRM  5216       // radial MLP trunk params : 4*336
#define F_NIX  6560       // neighbor idx (as int) : 16
#define F_TOT  6576
#define LDS_BYTES (U_BYTES + F_TOT * 4)   // ~202 KB < 320 KB/WGP

// k index held by (lane, elem) in the 16-bit WMMA A/B operand layout (wave32)
__device__ __forceinline__ int kmap(int lane, int e) {
  return e + ((e < 8) ? ((lane < 16) ? 0 : 8) : ((lane < 16) ? 8 : 16));
}
// inverse: ushort index inside a 512-element A tile for (row m, k)
__device__ __forceinline__ int a_idx(int m, int k) {
  int half, elem;
  if (k < 8)       { half = 0; elem = k;      }
  else if (k < 16) { half = 1; elem = k - 8;  }
  else if (k < 24) { half = 0; elem = k - 8;  }
  else             { half = 1; elem = k - 16; }
  return (half * 16 + m) * 16 + elem;
}
__device__ __forceinline__ unsigned short f2bf(float f) {  // RNE f32 -> bf16
  union { float f; unsigned u; } v; v.f = f;
  return (unsigned short)((v.u + 0x7fffu + ((v.u >> 16) & 1u)) >> 16);
}
__device__ __forceinline__ float silu_(float x) { return x / (1.f + expf(-x)); }

// ---- immediate-offset ds_swizzle xor butterfly (no index VALU, no clamping) ----
template <int XORM>
__device__ __forceinline__ float swz_xor(float x) {
  union { float f; int i; } u; u.f = x;
  u.i = __builtin_amdgcn_ds_swizzle(u.i, (XORM << 10) | 0x1f);
  return u.f;
}
// pack-reduce step: low lanes (lane&XORM==0) end with x_lo+x_hi, high with y_lo+y_hi
template <int XORM>
__device__ __forceinline__ float pkred(float x, float y, int lane) {
  float sel = (lane & XORM) ? x : y;
  float s = swz_xor<XORM>(sel);
  return (lane & XORM) ? (y + s) : (x + s);
}
// 8 independent sums over each 16-lane group in 8 swizzles.
// On return, lane L holds the group sum of p[L&7].
__device__ __forceinline__ float reduce8(const float p[8], int lane) {
  float a0 = pkred<1>(p[0], p[1], lane);
  float a1 = pkred<1>(p[2], p[3], lane);
  float a2 = pkred<1>(p[4], p[5], lane);
  float a3 = pkred<1>(p[6], p[7], lane);
  float b0 = pkred<2>(a0, a1, lane);
  float b1 = pkred<2>(a2, a3, lane);
  float c  = pkred<4>(b0, b1, lane);
  c += swz_xor<8>(c);
  return c;
}

#define WMMA_BF16(A, B, C) \
  __builtin_amdgcn_wmma_f32_16x16x32_bf16(false, (A), false, (B), (short)0, (C), false, false)

struct KParams {
  const float *x0, *x1, *rel, *D1, *basis;
  const int *nidx; const unsigned char *mask;
  const float *wself0, *wself1, *wout0, *wout1;
  const float *rw1[4], *rb1[4], *rg1[4], *rw2[4], *rb2[4], *rg2[4];
  const float *rw3[4], *rb3[4];
  const float *ws; float *out;
};

// ---------------------------------------------------------------------------
// Pass 1: per-node xi/xj linear projections into workspace.
// ws: xi0p[2048*64] | xj0p[2048*64] | xi1p[2048*96] | xj1p[2048*96]
// ---------------------------------------------------------------------------
__global__ void __launch_bounds__(256) dtp_proj(const float* x0, const float* x1,
    const float* wxi0, const float* wxi1, const float* wxj0, const float* wxj1,
    float* ws) {
  __shared__ float sx0[64];
  __shared__ float sx1[96];
  const int bn = blockIdx.x, tid = threadIdx.x;
  if (tid < 64) sx0[tid] = x0[bn * 64 + tid];
  else if (tid < 160) sx1[tid - 64] = x1[bn * 96 + (tid - 64)];
  __syncthreads();
  float* xi0p = ws;
  float* xj0p = ws + 131072;
  float* xi1p = ws + 262144;
  float* xj1p = ws + 458752;
  for (int idx = tid; idx < 320; idx += 256) {
    if (idx < 128) {
      int c = idx & 63;
      const float* W = (idx < 64) ? wxi0 : wxj0;
      float s = 0.f;
      for (int d = 0; d < 64; ++d) s += sx0[d] * W[d * 64 + c];
      ((idx < 64) ? xi0p : xj0p)[bn * 64 + c] = s;
    } else {
      int j = idx - 128, c = j % 96, ch = c / 3, q = c % 3;
      const float* W = (j < 96) ? wxi1 : wxj1;
      float s = 0.f;
      for (int d = 0; d < 32; ++d) s += sx1[d * 3 + q] * W[d * 32 + ch];
      ((j < 96) ? xi1p : xj1p)[bn * 96 + ch * 3 + q] = s;
    }
  }
}

// ---------------------------------------------------------------------------
// Pass 2: main fused kernel. 256 blocks x 256 threads (8 wave32s).
// ---------------------------------------------------------------------------
__global__ void __launch_bounds__(256, 1) dtp_main(KParams P) {
  extern __shared__ __align__(128) unsigned char smem_raw[];
  unsigned short* U = (unsigned short*)smem_raw;
  float* F = (float*)(smem_raw + U_BYTES);
  int* iN = (int*)(F + F_NIX);

  const int tid  = threadIdx.x;
  const int lane = tid & 31;
  const int wave = tid >> 5;

  const float* xi0p = P.ws;
  const float* xj0p = P.ws + 131072;
  const float* xi1p = P.ws + 262144;
  const float* xj1p = P.ws + 458752;

  // basis11 collapses to two scalars per m: even-j -> g1r[l,m], odd-j -> g1r[l,2-m]
  float se[3], so[3];
  #pragma unroll
  for (int m = 0; m < 3; ++m) {
    se[m] = P.basis[m * 5 + 0] + P.basis[m * 5 + 2] + P.basis[m * 5 + 4];
    so[m] = P.basis[m * 5 + 1] + P.basis[m * 5 + 3];
  }

  // warm L2 for the big weight arrays
  if (lane == 0) {
    __builtin_prefetch(P.rw3[0] + wave * 4096, 0, 1);
    __builtin_prefetch(P.rw3[1] + wave * 2048, 0, 1);
    __builtin_prefetch(P.rw3[2] + wave * 2048, 0, 1);
    __builtin_prefetch(P.rw3[3] + wave * 1024, 0, 1);
  }

  // ---------------- weight staging: f32 -> bf16, WMMA-B lane layout ----------
  for (int idx = tid; idx < 32768; idx += 256) {              // WB00
    int o = idx >> 10, rem = idx & 1023, s = rem >> 9, ln = (rem >> 4) & 31, e = rem & 15;
    int t = ln & 15, k = kmap(ln, e);
    U[U_WB00 + idx] = f2bf(P.rw3[0][t * 2048 + o * 64 + s * 32 + k]);
  }
  for (int idx = tid; idx < 16384; idx += 256) {              // WB10
    int o = idx >> 9, ln = (idx >> 4) & 31, e = idx & 15;
    int t = ln & 15, k = kmap(ln, e);
    U[U_WB10 + idx] = f2bf(P.rw3[1][t * 1024 + o * 32 + k]);
  }
  for (int idx = tid; idx < 16384; idx += 256) {              // WB01
    int o = idx >> 10, s = (idx >> 9) & 1, ln = (idx >> 4) & 31, e = idx & 15;
    int t = ln & 15, k = kmap(ln, e);
    U[U_WB01 + idx] = f2bf(P.rw3[2][t * 1024 + o * 64 + s * 32 + k]);
  }
  for (int idx = tid; idx < 16384; idx += 256) {              // WB11 (+b3 row @k=16)
    int j = idx >> 9, ln = (idx >> 4) & 31, e = idx & 15;
    int cc = ln & 15, k = kmap(ln, e);
    int n = (j >> 1) * 32 + (j & 1) * 16 + cc;
    float v = (k < 16) ? P.rw3[3][k * 512 + n] : ((k == 16) ? P.rb3[3][n] : 0.f);
    U[U_WB11 + idx] = f2bf(v);
  }
  for (int idx = tid; idx < 2048; idx += 256) {               // BB00
    int tile = idx >> 9, ot = tile >> 1, s = tile & 1, ln = (idx >> 4) & 31, e = idx & 15;
    int o = ot * 16 + (ln & 15), k = kmap(ln, e);
    U[U_BB00 + idx] = f2bf(P.rb3[0][o * 64 + s * 32 + k]);
  }
  for (int idx = tid; idx < 1024; idx += 256) {               // BB10
    int ot = idx >> 9, ln = (idx >> 4) & 31, e = idx & 15;
    int o = ot * 16 + (ln & 15), k = kmap(ln, e);
    U[U_BB10 + idx] = f2bf(P.rb3[1][o * 32 + k]);
  }
  for (int idx = tid; idx < 1024; idx += 256) {               // BB01
    int s = idx >> 9, ln = (idx >> 4) & 31, e = idx & 15;
    int o = ln & 15, k = kmap(ln, e);
    U[U_BB01 + idx] = f2bf(P.rb3[2][o * 64 + s * 32 + k]);
  }
  for (int idx = tid; idx < 1344; idx += 256) {               // MLP trunk params (f32)
    int hd = idx / 336, r = idx % 336;
    float v;
    if      (r < 16)  v = P.rw1[hd][r];
    else if (r < 32)  v = P.rb1[hd][r - 16];
    else if (r < 48)  v = P.rg1[hd][r - 32];
    else if (r < 304) v = P.rw2[hd][r - 48];
    else if (r < 320) v = P.rb2[hd][r - 304];
    else              v = P.rg2[hd][r - 320];
    F[F_PRM + idx] = v;
  }
  __syncthreads();

  // ---------------- node loop ----------------
  for (int it = 0; it < NPB; ++it) {
    const int bn    = blockIdx.x * NPB + it;
    const int bbase = bn & ~(N_ - 1);          // batch base row for gathers

    // -- phase 1a: node scalars + zero C buffers --
    if (tid < 16) {
      F[F_MSK + tid] = P.mask[bn * 16 + tid] ? 1.f : 0.f;
      F[F_RD + tid]  = P.rel[bn * 16 + tid];
      iN[tid]        = P.nidx[bn * 16 + tid];
    } else if (tid < 160) {
      F[F_D1 + (tid - 16)] = P.D1[bn * 144 + (tid - 16)];
    }
    for (int idx = tid; idx < 2560; idx += 256) F[F_CB0 + idx] = 0.f;
    __syncthreads();

    // -- phase 1b: radial MLP trunk (waves 0-1) || gather+rotate (waves 2-7) --
    if (tid < 64) {
      int e = tid & 15, hd = tid >> 4;
      const float* prm = F + F_PRM + hd * 336;
      float d = F[F_RD + e];
      float h1[16], h2[16], mu = 0.f;
      #pragma unroll
      for (int t = 0; t < 16; ++t) { h1[t] = silu_(d * prm[t] + prm[16 + t]); mu += h1[t]; }
      mu *= (1.f / 16.f);
      float var = 0.f;
      #pragma unroll
      for (int t = 0; t < 16; ++t) { float c = h1[t] - mu; var += c * c; }
      float rs = rsqrtf(var * (1.f / 16.f) + EPS_);
      #pragma unroll
      for (int t = 0; t < 16; ++t) h1[t] = (h1[t] - mu) * rs * prm[32 + t];
      mu = 0.f;
      #pragma unroll
      for (int o = 0; o < 16; ++o) {
        float s = prm[304 + o];
        #pragma unroll
        for (int t = 0; t < 16; ++t) s += h1[t] * prm[48 + t * 16 + o];
        h2[o] = silu_(s); mu += h2[o];
      }
      mu *= (1.f / 16.f);
      var = 0.f;
      #pragma unroll
      for (int o = 0; o < 16; ++o) { float c = h2[o] - mu; var += c * c; }
      rs = rsqrtf(var * (1.f / 16.f) + EPS_);
      #pragma unroll
      for (int o = 0; o < 16; ++o) h2[o] = (h2[o] - mu) * rs * prm[320 + o];
      if (hd < 3) {
        #pragma unroll
        for (int t = 0; t < 16; ++t) F[F_HB + hd * 256 + e * 16 + t] = h2[t];
      } else {   // head 11: A operand, k=16 carries 1.0 to pick up b3 row in B
        #pragma unroll
        for (int k = 0; k < 16; ++k) U[U_AH11 + a_idx(e, k)] = f2bf(h2[k]);
        U[U_AH11 + a_idx(e, 16)] = f2bf(1.f);
        #pragma unroll
        for (int k = 17; k < 32; ++k) U[U_AH11 + a_idx(e, k)] = 0;
      }
    } else {
      if (tid == 64) {
        float s = 0.f;
        for (int e = 0; e < 16; ++e) s += F[F_MSK + e];
        F[F_DEN] = fmaxf(s, 1.f);
      }
      for (int idx = tid - 64; idx < 1024; idx += 192) {       // g0 -> AG0 (bf16)
        int e = idx >> 6, i = idx & 63;
        int nr = bbase + iN[e];
        float v = xj0p[nr * 64 + i] + xi0p[bn * 64 + i];
        U[U_AG0 + (i >> 5) * 512 + a_idx(e, i & 31)] = f2bf(v);
      }
      for (int idx = tid - 64; idx < 512; idx += 192) {        // g1 -> rotate -> AG1 + TB
        int e = idx >> 5, l = idx & 31;
        int nr = bbase + iN[e];
        float a0 = xj1p[nr * 96 + l * 3 + 0] + xi1p[bn * 96 + l * 3 + 0];
        float a1 = xj1p[nr * 96 + l * 3 + 1] + xi1p[bn * 96 + l * 3 + 1];
        float a2 = xj1p[nr * 96 + l * 3 + 2] + xi1p[bn * 96 + l * 3 + 2];
        const float* Dm = F + F_D1 + e * 9;
        float g[3];
        #pragma unroll
        for (int q = 0; q < 3; ++q)
          g[q] = a0 * Dm[0 * 3 + q] + a1 * Dm[1 * 3 + q] + a2 * Dm[2 * 3 + q];
        U[U_AG1 + a_idx(e, l)] = f2bf(g[1]);
        #pragma unroll
        for (int m = 0; m < 3; ++m)
          F[F_TB + (e * 32 + l) * 3 + m] = se[m] * g[m] + so[m] * g[2 - m];
      }
    }
    __syncthreads();

    // ================= stage 2: WMMA GEMMs, specialized per head =============
    const int e0  = (lane & 16) ? 8 : 0;
    const int col = lane & 15;

    // ---- heads 00 & 01 + their bias tiles (A = g0, K=64) ----
    {
      v16bf a0 = *(const v16bf*)(U + U_AG0 + lane * 16);
      v16bf a1 = *(const v16bf*)(U + U_AG0 + 512 + lane * 16);
      float hm0[8], hm2[8];
      #pragma unroll
      for (int r = 0; r < 8; ++r) {
        hm0[r] = F[F_HB + 0 * 256 + (r + e0) * 16 + col];
        hm2[r] = F[F_HB + 2 * 256 + (r + e0) * 16 + col];
      }
      for (int o = wave; o < 32; o += 8) {                 // head 00
        v16bf b0 = *(const v16bf*)(U + U_WB00 + o * 1024 + lane * 16);
        v16bf b1 = *(const v16bf*)(U + U_WB00 + o * 1024 + 512 + lane * 16);
        v8f c = {};
        c = WMMA_BF16(a0, b0, c);
        c = WMMA_BF16(a1, b1, c);
        float p[8];
        #pragma unroll
        for (int r = 0; r < 8; ++r) p[r] = c[r] * hm0[r];
        float s = reduce8(p, lane);
        if (!(lane & 8)) atomicAdd(&F[F_CB0 + o * 16 + (lane & 7) + e0], s);
      }
      for (int o = wave; o < 16; o += 8) {                 // head 01 (m=1 slot)
        v16bf b0 = *(const v16bf*)(U + U_WB01 + o * 1024 + lane * 16);
        v16bf b1 = *(const v16bf*)(U + U_WB01 + o * 1024 + 512 + lane * 16);
        v8f c = {};
        c = WMMA_BF16(a0, b0, c);
        c = WMMA_BF16(a1, b1, c);
        float p[8];
        #pragma unroll
        for (int r = 0; r < 8; ++r) p[r] = c[r] * hm2[r];
        float s = reduce8(p, lane);
        if (!(lane & 8)) atomicAdd(&F[F_CB1 + (o * 3 + 1) * 16 + (lane & 7) + e0], s);
      }
      if (wave < 2) {                                      // head 00 bias (D col = o)
        v16bf b0 = *(const v16bf*)(U + U_BB00 + wave * 1024 + lane * 16);
        v16bf b1 = *(const v16bf*)(U + U_BB00 + wave * 1024 + 512 + lane * 16);
        v8f c = {};
        c = WMMA_BF16(a0, b0, c);
        c = WMMA_BF16(a1, b1, c);
        #pragma unroll
        for (int r = 0; r < 8; ++r)
          atomicAdd(&F[F_CB0 + (wave * 16 + col) * 16 + r + e0], c[r]);
      } else if (wave == 4) {                              // head 01 bias
        v16bf b0 = *(const v16bf*)(U + U_BB01 + lane * 16);
        v16bf b1 = *(const v16bf*)(U + U_BB01 + 512 + lane * 16);
        v8f c = {};
        c = WMMA_BF16(a0, b0, c);
        c = WMMA_BF16(a1, b1, c);
        #pragma unroll
        for (int r = 0; r < 8; ++r)
          atomicAdd(&F[F_CB1 + 16 + col * 48 + r + e0], c[r]);
      }
    }

    // ---- head 10 + bias (A = g1r[:,:,1], K=32) ----
    {
      v16bf a = *(const v16bf*)(U + U_AG1 + lane * 16);
      float hm1[8];
      #pragma unroll
      for (int r = 0; r < 8; ++r)
        hm1[r] = F[F_HB + 1 * 256 + (r + e0) * 16 + col];
      for (int o = wave; o < 32; o += 8) {
        v16bf b = *(const v16bf*)(U + U_WB10 + o * 512 + lane * 16);
        v8f c = {};
        c = WMMA_BF16(a, b, c);
        float p[8];
        #pragma unroll
        for (int r = 0; r < 8; ++r) p[r] = c[r] * hm1[r];
        float s = reduce8(p, lane);
        if (!(lane & 8)) atomicAdd(&F[F_CB0 + (32 + o) * 16 + (lane & 7) + e0], s);
      }
      if (wave == 2 || wave == 3) {                        // head 10 bias
        int ot = wave - 2;
        v16bf b = *(const v16bf*)(U + U_BB10 + ot * 512 + lane * 16);
        v8f c = {};
        c = WMMA_BF16(a, b, c);
        #pragma unroll
        for (int r = 0; r < 8; ++r)
          atomicAdd(&F[F_CB0 + (32 + ot * 16 + col) * 16 + r + e0], c[r]);
      }
    }

    // ---- head 11, R-form with b3 folded in @k=16 (A = h11 pad, K=32) ----
    {
      v16bf a = *(const v16bf*)(U + U_AH11 + lane * 16);
      // j stride of 8 keeps parity: i-base = (wave&1)*16 is loop-invariant
      float tm[3][8];
      #pragma unroll
      for (int m = 0; m < 3; ++m)
        #pragma unroll
        for (int r = 0; r < 8; ++r)
          tm[m][r] = F[F_TB + ((r + e0) * 32 + (wave & 1) * 16 + col) * 3 + m];
      for (int j = wave; j < 32; j += 8) {
        v16bf b = *(const v16bf*)(U + U_WB11 + j * 512 + lane * 16);
        v8f c = {};
        c = WMMA_BF16(a, b, c);
        const int oo = F_CB1 + ((16 + (j >> 1)) * 3) * 16;
        #pragma unroll
        for (int m = 0; m < 3; ++m) {
          float p[8];
          #pragma unroll
          for (int r = 0; r < 8; ++r) p[r] = c[r] * tm[m][r];
          float s = reduce8(p, lane);
          if (!(lane & 8)) atomicAdd(&F[oo + m * 16 + (lane & 7) + e0], s);
        }
      }
    }
    __syncthreads();

    // -- stage 3: masked mean pool (+ rotate deg-1 back out of edge frame) --
    if (tid < 64) {
      float s = 0.f;
      #pragma unroll
      for (int e = 0; e < 16; ++e) s += F[F_CB0 + tid * 16 + e] * F[F_MSK + e];
      F[F_P0 + tid] = s / F[F_DEN];
    } else if (tid < 160) {
      int idx = tid - 64, l = idx / 3, q = idx % 3;
      float s = 0.f;
      for (int e = 0; e < 16; ++e) {
        const float* Dm = F + F_D1 + e * 9;
        float acc = 0.f;
        #pragma unroll
        for (int a = 0; a < 3; ++a) acc += F[F_CB1 + (l * 3 + a) * 16 + e] * Dm[q * 3 + a];
        s += acc * F[F_MSK + e];
      }
      F[F_P1 + l * 3 + q] = s / F[F_DEN];
    }
    __syncthreads();

    // -- stage 4: output projection + self-interaction residual --
    if (tid < 64) {
      float s = 0.f;
      for (int d = 0; d < 64; ++d)
        s += F[F_P0 + d] * P.wout0[d * 64 + tid]
           + P.x0[bn * 64 + d] * P.wself0[d * 64 + tid];
      P.out[bn * 64 + tid] = s;
    } else if (tid < 160) {
      int idx = tid - 64, ch = idx / 3, q = idx % 3;
      float s = 0.f;
      for (int d = 0; d < 32; ++d)
        s += F[F_P1 + d * 3 + q] * P.wout1[d * 32 + ch]
           + P.x1[bn * 96 + d * 3 + q] * P.wself1[d * 32 + ch];
      P.out[131072 + bn * 96 + ch * 3 + q] = s;
    }
  }
}

// ---------------------------------------------------------------------------
// d_in order assumed = setup_inputs() dict insertion order, params flattened
// in insertion order: x0,x1,nidx,mask,rel,D1,basis, xi0,xi1,xj0,xj1,
// self0,self1,out0,out1, then radial '00','10','01','11' each
// {w1,b1,g1,w2,b2,g2,w3,b3}.
// ---------------------------------------------------------------------------
extern "C" void kernel_launch(void* const* d_in, const int* in_sizes, int n_in,
                              void* d_out, int out_size, void* d_ws, size_t ws_size,
                              hipStream_t stream) {
  (void)in_sizes; (void)n_in; (void)out_size; (void)ws_size;
  const float* x0    = (const float*)d_in[0];
  const float* x1    = (const float*)d_in[1];
  const int*   nidx  = (const int*)d_in[2];
  const unsigned char* mask = (const unsigned char*)d_in[3];  // jax bool -> 1B
  const float* rel   = (const float*)d_in[4];
  const float* D1g   = (const float*)d_in[5];
  const float* basis = (const float*)d_in[6];
  float* ws = (float*)d_ws;

  dtp_proj<<<NODES, 256, 0, stream>>>(x0, x1,
      (const float*)d_in[7], (const float*)d_in[8],
      (const float*)d_in[9], (const float*)d_in[10], ws);

  KParams P;
  P.x0 = x0; P.x1 = x1; P.rel = rel; P.D1 = D1g; P.basis = basis;
  P.nidx = nidx; P.mask = mask;
  P.wself0 = (const float*)d_in[11]; P.wself1 = (const float*)d_in[12];
  P.wout0  = (const float*)d_in[13]; P.wout1  = (const float*)d_in[14];
  for (int h = 0; h < 4; ++h) {
    int b = 15 + h * 8;
    P.rw1[h] = (const float*)d_in[b + 0];
    P.rb1[h] = (const float*)d_in[b + 1];
    P.rg1[h] = (const float*)d_in[b + 2];
    P.rw2[h] = (const float*)d_in[b + 3];
    P.rb2[h] = (const float*)d_in[b + 4];
    P.rg2[h] = (const float*)d_in[b + 5];
    P.rw3[h] = (const float*)d_in[b + 6];
    P.rb3[h] = (const float*)d_in[b + 7];
  }
  P.ws = ws; P.out = (float*)d_out;

  static_assert(LDS_BYTES < 320 * 1024, "LDS budget exceeded");
  hipFuncSetAttribute(reinterpret_cast<const void*>(dtp_main),
                      hipFuncAttributeMaxDynamicSharedMemorySize, LDS_BYTES);
  dtp_main<<<NODES / NPB, 256, LDS_BYTES, stream>>>(P);
}
